// Head_16561393893758
// MI455X (gfx1250) — compile-verified, matrix operand order checked by reading, not run on hardware
//
#include <hip/hip_runtime.h>

// One head of causal self-attention for MI455X (gfx1250, wave32, WMMA f16,
// async global->LDS staging, double-buffered, conflict-free padded LDS).
// B=32, T=2048, E=384, H=64.

typedef __attribute__((ext_vector_type(16))) _Float16 v16h;
typedef __attribute__((ext_vector_type(8)))  _Float16 v8h;
typedef __attribute__((ext_vector_type(8)))  float    v8f;

#define B_ 32
#define T_ 2048
#define E_ 384
#define H_ 64

#define LDM_KV 72   // padded LDS row stride (halves) for K/V tiles: 36 dwords -> no bank conflicts
#define LDM_P  40   // padded LDS row stride (halves) for P tile: 20 dwords -> no bank conflicts

#define WMMA_F16(A, Bm, C) \
  __builtin_amdgcn_wmma_f32_16x16x32_f16(false, (A), false, (Bm), (short)0, (C), false, false)

// 16-bit 16x32 A fragment (ISA 7.12.2): lane m = row; per lane the K indices are
// two contiguous 8-element runs:  hf*8 + [0,8)  and  16 + hf*8 + [0,8).

// A fragment from f16 row-major tile (global or LDS) -- two 16B vector loads.
static __device__ __forceinline__ v16h loadA_f16(const _Float16* base, int ldm, int row0, int col0) {
  int l = threadIdx.x & 31;
  int m = l & 15, hf = (l >> 4) & 1;
  const _Float16* p = base + (long)(row0 + m) * ldm + col0 + hf * 8;
  v8h lo = *(const v8h*)(p);
  v8h hi = *(const v8h*)(p + 16);
  v16h a;
#pragma unroll
  for (int i = 0; i < 8; ++i) { a[i] = lo[i]; a[8 + i] = hi[i]; }
  return a;
}

// A fragment from fp32 row-major source -- four float4 loads + convert.
static __device__ __forceinline__ v16h loadA_f32(const float* base, int ldm, int row0, int col0) {
  int l = threadIdx.x & 31;
  int m = l & 15, hf = (l >> 4) & 1;
  const float* p = base + (long)(row0 + m) * ldm + col0 + hf * 8;
  float4 f0 = *(const float4*)(p);
  float4 f1 = *(const float4*)(p + 4);
  float4 f2 = *(const float4*)(p + 16);
  float4 f3 = *(const float4*)(p + 20);
  v16h a;
  a[0]  = (_Float16)f0.x; a[1]  = (_Float16)f0.y; a[2]  = (_Float16)f0.z; a[3]  = (_Float16)f0.w;
  a[4]  = (_Float16)f1.x; a[5]  = (_Float16)f1.y; a[6]  = (_Float16)f1.z; a[7]  = (_Float16)f1.w;
  a[8]  = (_Float16)f2.x; a[9]  = (_Float16)f2.y; a[10] = (_Float16)f2.z; a[11] = (_Float16)f2.w;
  a[12] = (_Float16)f3.x; a[13] = (_Float16)f3.y; a[14] = (_Float16)f3.z; a[15] = (_Float16)f3.w;
  return a;
}

// B fragment (32x16) from fp32 row-major source: element (k, n) -- column gather.
static __device__ __forceinline__ v16h loadB_f32(const float* base, int ldm, int k0, int n0) {
  int l = threadIdx.x & 31;
  int n = l & 15, hf = (l >> 4) & 1;
  v16h b;
#pragma unroll
  for (int g = 0; g < 8; ++g) {
    int k = ((g & 4) ? 16 : 0) + hf * 8 + 2 * (g & 3);
    b[2 * g]     = (_Float16)base[(long)(k0 + k) * ldm + n0 + n];
    b[2 * g + 1] = (_Float16)base[(long)(k0 + k + 1) * ldm + n0 + n];
  }
  return b;
}

// B fragment (32x16) from an f16 row-major tile: element (k, n) -- column gather.
static __device__ __forceinline__ v16h loadB_f16(const _Float16* base, int ldm, int k0, int n0) {
  int l = threadIdx.x & 31;
  int n = l & 15, hf = (l >> 4) & 1;
  v16h b;
#pragma unroll
  for (int g = 0; g < 8; ++g) {
    int k = ((g & 4) ? 16 : 0) + hf * 8 + 2 * (g & 3);
    b[2 * g]     = base[(long)(k0 + k) * ldm + n0 + n];
    b[2 * g + 1] = base[(long)(k0 + k + 1) * ldm + n0 + n];
  }
  return b;
}

// Reductions across the 16 lanes holding one C-matrix row (xor masks <= 8 stay in-half).
static __device__ __forceinline__ float redMax16(float v) {
#pragma unroll
  for (int m = 1; m <= 8; m <<= 1) v = fmaxf(v, __shfl_xor(v, m, 32));
  return v;
}
static __device__ __forceinline__ float redSum16(float v) {
#pragma unroll
  for (int m = 1; m <= 8; m <<= 1) v += __shfl_xor(v, m, 32);
  return v;
}

// CDNA5 async DMA: global -> LDS, 16B per lane, tracked by ASYNCcnt.
// Low 32 bits of a generic shared-memory address are the LDS byte offset
// (ISA 10.2).
static __device__ __forceinline__ void async_cp_b128(_Float16* lds_dst, const _Float16* gsrc) {
  unsigned lds_off = (unsigned)(unsigned long long)lds_dst;
  asm volatile("global_load_async_to_lds_b128 %0, %1, off"
               :: "v"(lds_off), "v"(gsrc) : "memory");
}

// ---------------- Kernel 1: q/k/v = x @ {Wq,Wk,Wv}, fp32 -> f16, q pre-scaled --
// One wave per 16-row tile of x; A fragment loaded once, reused by 12 WMMAs.
__global__ __launch_bounds__(128) void qkv_proj_kernel(
    const float* __restrict__ x, const float* __restrict__ Wk,
    const float* __restrict__ Wq, const float* __restrict__ Wv,
    _Float16* __restrict__ qb, _Float16* __restrict__ kb, _Float16* __restrict__ vb) {
  int wave = blockIdx.x * (blockDim.x >> 5) + (threadIdx.x >> 5);
  int row0 = wave * 16;                 // 16-row tile of flattened [B*T, E]

  const float* Ws[3] = {Wq, Wk, Wv};
  v8f c[12] = {};                        // [mat][ntile] accumulators

  for (int kc = 0; kc < E_; kc += 32) {
    v16h a = loadA_f32(x, E_, row0, kc);
#pragma unroll
    for (int m = 0; m < 3; ++m) {
#pragma unroll
      for (int nt = 0; nt < 4; ++nt) {
        v16h b = loadB_f32(Ws[m], H_, kc, nt * 16);
        c[m * 4 + nt] = WMMA_F16(a, b, c[m * 4 + nt]);
      }
    }
  }

  int l = threadIdx.x & 31;
  int n = l & 15, hf = (l >> 4) & 1;
  _Float16* outs[3] = {qb, kb, vb};
  float scls[3] = {0.125f, 1.0f, 1.0f};  // fold 1/sqrt(H) into q
#pragma unroll
  for (int m = 0; m < 3; ++m) {
#pragma unroll
    for (int nt = 0; nt < 4; ++nt) {
#pragma unroll
      for (int r = 0; r < 8; ++r) {
        long row = row0 + r + hf * 8;
        outs[m][row * H_ + nt * 16 + n] = (_Float16)(c[m * 4 + nt][r] * scls[m]);
      }
    }
  }
}

// ---------------- Kernel 2: causal flash attention ----------------------------
// Block = 4 waves = 4 consecutive 16-query tiles (same batch).  K/V tiles of 32
// keys are staged cooperatively into LDS with async b128 copies, double-buffered.
__global__ __launch_bounds__(128) void flash_attn_kernel(
    const _Float16* __restrict__ qb, const _Float16* __restrict__ kb,
    const _Float16* __restrict__ vb, float* __restrict__ out) {
  __shared__ __align__(16) _Float16 Kt[2][32][LDM_KV];   // padded: conflict-free b128
  __shared__ __align__(16) _Float16 Vt[2][32][LDM_KV];
  __shared__ __align__(16) _Float16 Pb[4][16 * LDM_P];   // per-wave P bounce

  int wslot = threadIdx.x >> 5;
  int tile  = blockIdx.x * 4 + wslot;        // [0, B*T/16); 4 tiles share a batch
  int b     = tile / (T_ / 16);
  int q0    = (tile % (T_ / 16)) * 16;
  int lt0   = (blockIdx.x * 4) % (T_ / 16);  // first tile-in-batch of this block
  int nsteps = (lt0 * 16 + 64) / 32;         // covers causal range of all 4 waves

  int l = threadIdx.x & 31;
  int nl = l & 15, hf = (l >> 4) & 1;

  const _Float16* Qp = qb + (long)b * T_ * H_;
  const _Float16* Kp = kb + (long)b * T_ * H_;
  const _Float16* Vp = vb + (long)b * T_ * H_;

  // Q fragments stay resident: h-chunks [0,32) and [32,64)
  v16h qa0 = loadA_f16(Qp, H_, q0, 0);
  v16h qa1 = loadA_f16(Qp, H_, q0, 32);

  v8f o0 = {}, o1 = {}, o2 = {}, o3 = {};
  float rowM[8], rowL[8];
#pragma unroll
  for (int r = 0; r < 8; ++r) { rowM[r] = -3.0e38f; rowL[r] = 0.0f; }

  _Float16* P = Pb[wslot];
  const int kend = q0 + 16;                  // this wave needs keys < kend

  // Stage step 0 into buffer 0: 512 x b128 over 128 threads = 4 per thread.
  int tid = threadIdx.x;
#pragma unroll
  for (int j = 0; j < 4; ++j) {
    int t  = tid * 4 + j;                    // 0..511
    int isV = t >> 8, r = (t & 255) >> 3, sg = t & 7;
    const _Float16* src = (isV ? Vp : Kp) + (long)r * H_ + sg * 8;
    _Float16* dst = (isV ? &Vt[0][0][0] : &Kt[0][0][0]) + r * LDM_KV + sg * 8;
    async_cp_b128(dst, src);
  }

  for (int i = 0; i < nsteps; ++i) {
    int kt = i * 32;
    int cur = i & 1;
    __syncthreads();                         // buf[1^cur] no longer being read
    bool pf = (i + 1 < nsteps);
    if (pf) {
#pragma unroll
      for (int j = 0; j < 4; ++j) {
        int t  = tid * 4 + j;
        int isV = t >> 8, r = (t & 255) >> 3, sg = t & 7;
        const _Float16* src = (isV ? Vp : Kp) + (long)(kt + 32 + r) * H_ + sg * 8;
        _Float16* dst = (isV ? &Vt[1 ^ cur][0][0] : &Kt[1 ^ cur][0][0]) + r * LDM_KV + sg * 8;
        async_cp_b128(dst, src);
      }
      asm volatile("s_wait_asynccnt 0x4" ::: "memory");  // cur tile landed
    } else {
      asm volatile("s_wait_asynccnt 0x0" ::: "memory");
    }
    __syncthreads();                         // cur tile visible block-wide

    if (kt < kend) {
      const _Float16* Kl = &Kt[cur][0][0];
      const _Float16* Vl = &Vt[cur][0][0];

      // ---- S = Q K^T (scores pre-scaled via q): 16q x 32k as two fragments
      v8f s0 = {}, s1 = {};
      {
        v16h t0 = loadA_f16(Kl, LDM_KV, 0, 0);
        v16h t1 = loadA_f16(Kl, LDM_KV, 0, 32);
        s0 = WMMA_F16(qa0, t0, s0);
        s0 = WMMA_F16(qa1, t1, s0);
        v16h t2 = loadA_f16(Kl, LDM_KV, 16, 0);
        v16h t3 = loadA_f16(Kl, LDM_KV, 16, 32);
        s1 = WMMA_F16(qa0, t2, s1);
        s1 = WMMA_F16(qa1, t3, s1);
      }

      // ---- causal mask (diagonal tiles only)
      if (kt + 31 > q0) {
#pragma unroll
        for (int r = 0; r < 8; ++r) {
          int qrow = q0 + r + hf * 8;
          if (kt + nl      > qrow) s0[r] = -3.0e38f;
          if (kt + 16 + nl > qrow) s1[r] = -3.0e38f;
        }
      }

      // ---- online softmax; write P (f16) to per-wave LDS in natural layout
      float scale[8];
#pragma unroll
      for (int r = 0; r < 8; ++r) {
        float rm = redMax16(fmaxf(s0[r], s1[r]));
        float m2 = fmaxf(rowM[r], rm);
        float sc = __expf(rowM[r] - m2);
        float p0 = __expf(s0[r] - m2);
        float p1 = __expf(s1[r] - m2);
        int row = r + hf * 8;
        P[row * LDM_P + nl]      = (_Float16)p0;
        P[row * LDM_P + 16 + nl] = (_Float16)p1;
        rowL[r] = rowL[r] * sc + redSum16(p0 + p1);
        rowM[r] = m2;
        scale[r] = sc;
      }
#pragma unroll
      for (int r = 0; r < 8; ++r) {
        o0[r] *= scale[r]; o1[r] *= scale[r];
        o2[r] *= scale[r]; o3[r] *= scale[r];
      }

      // C-layout -> A-layout transpose via LDS (same-wave DS ops are in-order)
      asm volatile("s_wait_dscnt 0x0" ::: "memory");
      v16h pa = loadA_f16(P, LDM_P, 0, 0);

      // ---- O += P V  (V from LDS: rows = 32 keys, cols = 64 head dims)
      o0 = WMMA_F16(pa, loadB_f16(Vl, LDM_KV, 0, 0),  o0);
      o1 = WMMA_F16(pa, loadB_f16(Vl, LDM_KV, 0, 16), o1);
      o2 = WMMA_F16(pa, loadB_f16(Vl, LDM_KV, 0, 32), o2);
      o3 = WMMA_F16(pa, loadB_f16(Vl, LDM_KV, 0, 48), o3);
    }
  }

  // ---- normalize and store fp32 output [B, T, H]
  float* Op = out + (long)b * T_ * H_;
#pragma unroll
  for (int r = 0; r < 8; ++r) {
    float inv = 1.0f / rowL[r];
    long row = q0 + r + hf * 8;
    Op[row * H_ + 0  + nl] = o0[r] * inv;
    Op[row * H_ + 16 + nl] = o1[r] * inv;
    Op[row * H_ + 32 + nl] = o2[r] * inv;
    Op[row * H_ + 48 + nl] = o3[r] * inv;
  }
}

extern "C" void kernel_launch(void* const* d_in, const int* in_sizes, int n_in,
                              void* d_out, int out_size, void* d_ws, size_t ws_size,
                              hipStream_t stream) {
  const float* x  = (const float*)d_in[0];
  const float* Wk = (const float*)d_in[1];
  const float* Wq = (const float*)d_in[2];
  const float* Wv = (const float*)d_in[3];
  float* out = (float*)d_out;

  const long N = (long)B_ * T_ * H_;        // 4,194,304 elements
  _Float16* qbuf = (_Float16*)d_ws;         // 3*N*2 = ~25 MB of workspace
  _Float16* kbuf = qbuf + N;
  _Float16* vbuf = kbuf + N;

  {
    int rowTiles = (B_ * T_) / 16;          // 4096 waves, one per row tile
    qkv_proj_kernel<<<rowTiles / 4, 128, 0, stream>>>(x, Wk, Wq, Wv, qbuf, kbuf, vbuf);
  }
  {
    int qtiles = (B_ * T_) / 16;            // 4096 q-tiles, 4 per block
    flash_attn_kernel<<<qtiles / 4, 128, 0, stream>>>(qbuf, kbuf, vbuf, out);
  }
}
